// MantaFrontierPredictor_59785944760881
// MI455X (gfx1250) — compile-verified, matrix operand order checked by reading, not run on hardware
//
#include <hip/hip_runtime.h>
#include <cstdint>
#include <cstddef>

#define Bz 2
#define Sq 4096
#define HIDN 768
#define NLAYERS 4
#define NHEADS 12
#define HDIM 64
#define WH 128
#define NCHUNK 32
#define FFN 3072
#define MTOK (Bz*Sq)
#define SLDS 388   // 384 + 4 pad to spread LDS banks (attention scores)
#define SROW 40    // 32 + 8 pad halves -> 80B pitch, conflict-free b128 reads

typedef _Float16 half8  __attribute__((ext_vector_type(8)));
typedef _Float16 half16 __attribute__((ext_vector_type(16)));
typedef float    float8 __attribute__((ext_vector_type(8)));
typedef int      mfp_v4i __attribute__((vector_size(16)));

// ---- CDNA5 async global->LDS (ASYNCcnt-tracked). Builtin prototype (from
// hipcc diagnostic): (v4i32 addrspace(1)*, v4i32 addrspace(3)*, imm, imm).
#if defined(__HIP_DEVICE_COMPILE__)
#if __has_builtin(__builtin_amdgcn_global_load_async_to_lds_b128)
#define MFP_ASYNC_B128(gp, lp)                                                  \
  __builtin_amdgcn_global_load_async_to_lds_b128(                               \
      (__attribute__((address_space(1))) mfp_v4i*)(size_t)(gp),                 \
      (__attribute__((address_space(3))) mfp_v4i*)(unsigned)(size_t)(lp), 0, 0)
#else
#define MFP_ASYNC_B128(gp, lp)                                                  \
  asm volatile("global_load_async_to_lds_b128 %0, %1, off"                      \
               :: "v"((unsigned)(size_t)(lp)),                                  \
                  "v"((unsigned long long)(size_t)(gp)) : "memory")
#endif
#if __has_builtin(__builtin_amdgcn_s_wait_asynccnt)
#define MFP_WAIT_ASYNC(n) __builtin_amdgcn_s_wait_asynccnt(n)
#else
#define MFP_WAIT_ASYNC(n) asm volatile("s_wait_asynccnt %0" :: "i"(n) : "memory")
#endif
#else
#define MFP_ASYNC_B128(gp, lp) ((void)0)
#define MFP_WAIT_ASYNC(n)      ((void)0)
#endif

// ---------------------------------------------------------------- helpers

__device__ __forceinline__ half16 mfp_combine(half8 lo, half8 hi) {
  half16 r;
#pragma unroll
  for (int i = 0; i < 8; ++i) { r[i] = lo[i]; r[i + 8] = hi[i]; }
  return r;
}

__device__ __forceinline__ float8 mfp_wmma(half16 a, half16 b, float8 c) {
  // D = A(16x32) * B(32x16) + C, f32 accumulate
  return __builtin_amdgcn_wmma_f32_16x16x32_f16(false, a, false, b,
                                                (short)0, c, false, false);
}

// CDNA5 16-bit fragment: lane covers row/col = base + (lane&15); element e
// maps to k = k0 + (e<8 ? e : e+8) + (lane>=16 ? 8 : 0) -> two contiguous
// 16B half8 loads per lane.
__device__ __forceinline__ half16 mfp_ldfrag(const _Float16* __restrict__ p,
                                             int ld, int row, int k0) {
  int lane = threadIdx.x & 31;
  int sel  = lane >> 4;
  const _Float16* base = p + (size_t)(row + (lane & 15)) * ld + k0 + sel * 8;
  half8 lo = *reinterpret_cast<const half8*>(base);
  half8 hi = *reinterpret_cast<const half8*>(base + 16);
  return mfp_combine(lo, hi);
}

// Fragment out of an LDS-staged tile (pitch SROW halves, 32 valid k per row).
__device__ __forceinline__ half16 mfp_ldfrag_s(const _Float16* p, int row) {
  int lane = threadIdx.x & 31;
  int sel  = lane >> 4;
  const _Float16* base = p + (row + (lane & 15)) * SROW + sel * 8;
  half8 lo = *reinterpret_cast<const half8*>(base);
  half8 hi = *reinterpret_cast<const half8*>(base + 16);
  return mfp_combine(lo, hi);
}

// B fragment for scores: col = key-window index j, k = head dim (contiguous).
__device__ __forceinline__ half16 mfp_ldfrag_keys(const _Float16* __restrict__ base,
                                                  int c, int jbase, int k0) {
  int lane = threadIdx.x & 31;
  int sel  = lane >> 4;
  int j    = jbase + (lane & 15);
  int t    = c * WH + j - WH;
  t = t < 0 ? 0 : (t > Sq - 1 ? Sq - 1 : t);
  const _Float16* p = base + (size_t)t * HIDN + k0 + sel * 8;
  half8 lo = *reinterpret_cast<const half8*>(p);
  half8 hi = *reinterpret_cast<const half8*>(p + 16);
  return mfp_combine(lo, hi);
}

// A fragment (probs) out of f32 LDS scores, converted to f16.
__device__ __forceinline__ half16 mfp_ldfrag_lds(const float* base, int ld, int k0) {
  int lane = threadIdx.x & 31;
  int sel  = lane >> 4;
  const float* p = base + (lane & 15) * ld + k0 + sel * 8;
  half16 o;
#pragma unroll
  for (int i = 0; i < 8; ++i) { o[i] = (_Float16)p[i]; o[i + 8] = (_Float16)p[i + 16]; }
  return o;
}

// B fragment for out = P @ V: col = head dim d, element k = key index j.
__device__ __forceinline__ half16 mfp_ldfrag_v(const _Float16* __restrict__ vbase,
                                               int dbase, int c, int k0) {
  int lane = threadIdx.x & 31;
  int sel  = lane >> 4;
  int d    = dbase + (lane & 15);
  half16 o;
#pragma unroll
  for (int e = 0; e < 16; ++e) {
    int k = k0 + ((e < 8) ? e : e + 8) + sel * 8;
    int t = c * WH + k - WH;
    t = t < 0 ? 0 : (t > Sq - 1 ? Sq - 1 : t);
    o[e] = vbase[(size_t)t * HIDN + d];
  }
  return o;
}

// ---------------------------------------------------------------- kernels

__global__ __launch_bounds__(256) void mfp_f2h(const float* __restrict__ in,
                                               _Float16* __restrict__ out, int n) {
  int i = blockIdx.x * 256 + threadIdx.x;
  if (i < n) out[i] = (_Float16)in[i];
}

// W [K,N] f32 -> Wt [N,K] f16, LDS-tiled 32x32 transpose (coalesced both sides)
__global__ __launch_bounds__(256) void mfp_wtrans(const float* __restrict__ Win,
                                                  _Float16* __restrict__ Wt,
                                                  int K, int N) {
  __shared__ _Float16 tile[32][33];
  int tk = blockIdx.y * 32, tn = blockIdx.x * 32;
  int c  = threadIdx.x & 31, r0 = threadIdx.x >> 5;
#pragma unroll
  for (int r = r0; r < 32; r += 8)
    tile[r][c] = (_Float16)Win[(size_t)(tk + r) * N + tn + c];
  __syncthreads();
#pragma unroll
  for (int r = r0; r < 32; r += 8)
    Wt[(size_t)(tn + r) * K + tk + c] = tile[c][r];
}

// h = LN(emb + pos[1+s] + tok[0]) * g + b       one block per token row
__global__ __launch_bounds__(256) void mfp_embed_ln(
    const float* __restrict__ emb, const float* __restrict__ pos,
    const float* __restrict__ tok, const float* __restrict__ g,
    const float* __restrict__ bt, float* __restrict__ h) {
  int row = blockIdx.x;
  int s   = row % Sq;
  int t   = threadIdx.x;
  __shared__ float red[256];
  float v[3];
#pragma unroll
  for (int i = 0; i < 3; ++i) {
    int idx = t + i * 256;
    v[i] = emb[(size_t)row * HIDN + idx] + pos[(size_t)(s + 1) * HIDN + idx] + tok[idx];
  }
  float loc = v[0] + v[1] + v[2];
  red[t] = loc; __syncthreads();
  for (int st = 128; st > 0; st >>= 1) { if (t < st) red[t] += red[t + st]; __syncthreads(); }
  float mean = red[0] * (1.0f / HIDN);
  __syncthreads();
  loc = 0.f;
#pragma unroll
  for (int i = 0; i < 3; ++i) { float d = v[i] - mean; loc += d * d; }
  red[t] = loc; __syncthreads();
  for (int st = 128; st > 0; st >>= 1) { if (t < st) red[t] += red[t + st]; __syncthreads(); }
  float rs = rsqrtf(red[0] * (1.0f / HIDN) + 1e-12f);
#pragma unroll
  for (int i = 0; i < 3; ++i) {
    int idx = t + i * 256;
    h[(size_t)row * HIDN + idx] = (v[i] - mean) * rs * g[idx] + bt[idx];
  }
}

// h = LN(h + x) * g + b      (in-place on h; one block per row)
__global__ __launch_bounds__(256) void mfp_add_ln(
    float* __restrict__ h, const float* __restrict__ x,
    const float* __restrict__ g, const float* __restrict__ bt) {
  int row = blockIdx.x;
  int t   = threadIdx.x;
  __shared__ float red[256];
  float v[3];
#pragma unroll
  for (int i = 0; i < 3; ++i) {
    int idx = t + i * 256;
    v[i] = h[(size_t)row * HIDN + idx] + x[(size_t)row * HIDN + idx];
  }
  float loc = v[0] + v[1] + v[2];
  red[t] = loc; __syncthreads();
  for (int st = 128; st > 0; st >>= 1) { if (t < st) red[t] += red[t + st]; __syncthreads(); }
  float mean = red[0] * (1.0f / HIDN);
  __syncthreads();
  loc = 0.f;
#pragma unroll
  for (int i = 0; i < 3; ++i) { float d = v[i] - mean; loc += d * d; }
  red[t] = loc; __syncthreads();
  for (int st = 128; st > 0; st >>= 1) { if (t < st) red[t] += red[t + st]; __syncthreads(); }
  float rs = rsqrtf(red[0] * (1.0f / HIDN) + 1e-12f);
#pragma unroll
  for (int i = 0; i < 3; ++i) {
    int idx = t + i * 256;
    h[(size_t)row * HIDN + idx] = (v[i] - mean) * rs * g[idx] + bt[idx];
  }
}

// C = A[M,K]f16 @ Wt[N,K]f16 + bias, optional exact GELU; optional f32 and/or
// scaled f16 outputs. Block = 8 waves -> 128x64 tile, wave -> 32x32 (2x2 WMMA).
// Tiles double-buffered in LDS via async global->LDS (ASYNCcnt), WMMA fed by
// conflict-free ds_load_b128 (80B pitch).
__global__ __launch_bounds__(256) void mfp_gemm(
    const _Float16* __restrict__ A, const _Float16* __restrict__ Wt,
    const float* __restrict__ bias, float* __restrict__ Cf,
    _Float16* __restrict__ Ch, int M, int N, int K, int act, float hscale) {
  __shared__ _Float16 sA[2][128][SROW];
  __shared__ _Float16 sB[2][64][SROW];
  int tid  = threadIdx.x;
  int lane = tid & 31;
  int wave = tid >> 5;
  int blockRow = blockIdx.y * 128;
  int blockCol = blockIdx.x * 64;
  int rowW = (wave & 3) * 32;
  int colW = (wave >> 2) * 32;
  // async staging: thread t copies A rows t/4 and 64+t/4, B row t/4, 16B each
  int ar = tid >> 2;                 // 0..63
  int ak = (tid & 3) * 8;            // 0,8,16,24 (halves)

  auto issue = [&](int buf, int k0) {
    const _Float16* ga0 = A  + (size_t)(blockRow + ar)      * K + k0 + ak;
    const _Float16* ga1 = A  + (size_t)(blockRow + 64 + ar) * K + k0 + ak;
    const _Float16* gb  = Wt + (size_t)(blockCol + ar)      * K + k0 + ak;
    MFP_ASYNC_B128(ga0, &sA[buf][ar][ak]);
    MFP_ASYNC_B128(ga1, &sA[buf][64 + ar][ak]);
    MFP_ASYNC_B128(gb,  &sB[buf][ar][ak]);
  };

  float8 acc[2][2] = {};
  int nk = K >> 5;
  issue(0, 0);
  for (int kb = 0; kb < nk; ++kb) {
    int cur = kb & 1;
    __syncthreads();                               // prev compute done: cur^1 free
    if (kb + 1 < nk) { issue(cur ^ 1, (kb + 1) * 32); MFP_WAIT_ASYNC(3); }
    else             { MFP_WAIT_ASYNC(0); }
    __syncthreads();                               // cur tile visible to all waves
    half16 a0 = mfp_ldfrag_s(&sA[cur][0][0], rowW);
    half16 a1 = mfp_ldfrag_s(&sA[cur][0][0], rowW + 16);
    half16 b0 = mfp_ldfrag_s(&sB[cur][0][0], colW);
    half16 b1 = mfp_ldfrag_s(&sB[cur][0][0], colW + 16);
    acc[0][0] = mfp_wmma(a0, b0, acc[0][0]);
    acc[0][1] = mfp_wmma(a0, b1, acc[0][1]);
    acc[1][0] = mfp_wmma(a1, b0, acc[1][0]);
    acc[1][1] = mfp_wmma(a1, b1, acc[1][1]);
  }
  int ccol = lane & 15;
  int r0   = (lane >> 4) * 8;
#pragma unroll
  for (int fm = 0; fm < 2; ++fm)
#pragma unroll
    for (int fn = 0; fn < 2; ++fn) {
      int col = blockCol + colW + fn * 16 + ccol;
      float bb = bias ? bias[col] : 0.0f;
#pragma unroll
      for (int r = 0; r < 8; ++r) {
        int row = blockRow + rowW + fm * 16 + r0 + r;
        float v = acc[fm][fn][r] + bb;
        if (act == 1) v = 0.5f * v * (1.0f + erff(v * 0.70710678118654752f));
        size_t o = (size_t)row * N + col;
        if (Cf) Cf[o] = v;
        if (Ch) Ch[o] = (_Float16)(v * hscale);
      }
    }
}

// Sliding-window attention. One block (2 waves) per (b, head, chunk, 32-row
// tile); each wave owns 16 query rows. q16 is pre-scaled by 1/sqrt(64).
__global__ __launch_bounds__(64) void mfp_attn(
    const _Float16* __restrict__ q16, const _Float16* __restrict__ k16,
    const _Float16* __restrict__ v16, const int* __restrict__ amask,
    float* __restrict__ attn) {
  __shared__ float sc[2][16][SLDS];
  int lane = threadIdx.x & 31;
  int w    = threadIdx.x >> 5;
  int idx  = blockIdx.x;                 // b*1536 + hh*128 + c*4 + rt
  int rt   = idx & 3;
  int c    = (idx >> 2) & 31;
  int hh   = (idx >> 7) % NHEADS;
  int b    = idx / (4 * NCHUNK * NHEADS);
  int qrow0 = c * WH + rt * 32 + w * 16;
  const _Float16* qb = q16 + (size_t)b * Sq * HIDN + hh * HDIM;
  const _Float16* kb = k16 + (size_t)b * Sq * HIDN + hh * HDIM;
  const _Float16* vb = v16 + (size_t)b * Sq * HIDN + hh * HDIM;

  // phase 1: masked scores -> LDS
  for (int jf = 0; jf < 24; ++jf) {
    int jbase = jf * 16;
    float8 acc = {};
    for (int k0 = 0; k0 < HDIM; k0 += 32) {
      half16 a  = mfp_ldfrag(qb, HIDN, qrow0, k0);
      half16 bf = mfp_ldfrag_keys(kb, c, jbase, k0);
      acc = mfp_wmma(a, bf, acc);
    }
    int j = jbase + (lane & 15);
    int t = c * WH + j - WH;
    bool kval = false;
    if (t >= 0 && t < Sq) kval = (amask[(size_t)b * Sq + t] != 0);
#pragma unroll
    for (int r = 0; r < 8; ++r) {
      int lr = (lane >> 4) * 8 + r;            // row in 16-row tile
      int p  = rt * 32 + w * 16 + lr;          // query index within chunk
      bool band = (j >= p) && (j <= p + 2 * WH);
      sc[w][lr][j] = (kval && band) ? acc[r] : -1e9f;
    }
  }
  __syncthreads();

  // phase 2: softmax per row
  if (lane < 16) {
    float* row = sc[w][lane];
    float m = -1e30f;
    for (int j = 0; j < 384; ++j) m = fmaxf(m, row[j]);
    float s = 0.f;
    for (int j = 0; j < 384; ++j) { float e = expf(row[j] - m); row[j] = e; s += e; }
    float inv = 1.0f / s;
    for (int j = 0; j < 384; ++j) row[j] *= inv;
  }
  __syncthreads();

  // phase 3: out = P(16x384) @ V(384x64)
  float8 oacc[4] = {};
  for (int k0 = 0; k0 < 384; k0 += 32) {
    half16 a = mfp_ldfrag_lds(&sc[w][0][0], SLDS, k0);
#pragma unroll
    for (int fn = 0; fn < 4; ++fn) {
      half16 bf = mfp_ldfrag_v(vb, fn * 16, c, k0);
      oacc[fn] = mfp_wmma(a, bf, oacc[fn]);
    }
  }
  int ccol = lane & 15;
  int r0   = (lane >> 4) * 8;
#pragma unroll
  for (int fn = 0; fn < 4; ++fn)
#pragma unroll
    for (int r = 0; r < 8; ++r) {
      int q = qrow0 + r0 + r;
      attn[((size_t)b * Sq + q) * HIDN + hh * HDIM + fn * 16 + ccol] = oacc[fn][r];
    }
}

// out = sigmoid(h . Wp + bp)      one block per token row
__global__ __launch_bounds__(256) void mfp_head(
    const float* __restrict__ h, const float* __restrict__ Wp,
    const float* __restrict__ bp, float* __restrict__ out) {
  int row = blockIdx.x;
  int t   = threadIdx.x;
  __shared__ float red[256];
  float loc = 0.f;
#pragma unroll
  for (int i = 0; i < 3; ++i) {
    int idx = t + i * 256;
    loc += h[(size_t)row * HIDN + idx] * Wp[idx];
  }
  red[t] = loc; __syncthreads();
  for (int st = 128; st > 0; st >>= 1) { if (t < st) red[t] += red[t + st]; __syncthreads(); }
  if (t == 0) out[row] = 1.0f / (1.0f + expf(-(red[0] + bp[0])));
}

// ---------------------------------------------------------------- driver

extern "C" void kernel_launch(void* const* d_in, const int* in_sizes, int n_in,
                              void* d_out, int out_size, void* d_ws, size_t ws_size,
                              hipStream_t stream) {
  const float* emb  = (const float*)d_in[0];
  const int*   am   = (const int*)  d_in[1];
  const float* pos  = (const float*)d_in[2];
  const float* tok  = (const float*)d_in[3];
  const float* elng = (const float*)d_in[4];
  const float* elnb = (const float*)d_in[5];
  const float* Wq = (const float*)d_in[6];  const float* bq = (const float*)d_in[7];
  const float* Wk = (const float*)d_in[8];  const float* bk = (const float*)d_in[9];
  const float* Wv = (const float*)d_in[10]; const float* bv = (const float*)d_in[11];
  const float* Wo = (const float*)d_in[12]; const float* bo = (const float*)d_in[13];
  const float* l1g = (const float*)d_in[14]; const float* l1b = (const float*)d_in[15];
  const float* Wi = (const float*)d_in[16]; const float* bi = (const float*)d_in[17];
  const float* Wf = (const float*)d_in[18]; const float* bf = (const float*)d_in[19];
  const float* l2g = (const float*)d_in[20]; const float* l2b = (const float*)d_in[21];
  const float* Wp = (const float*)d_in[22]; const float* bp = (const float*)d_in[23];
  float* out = (float*)d_out;

  char* p = (char*)d_ws;
  float*    h    = (float*)p;     p += (size_t)MTOK * HIDN * 4;
  float*    tmp  = (float*)p;     p += (size_t)MTOK * FFN  * 4;
  float*    attn = (float*)p;     p += (size_t)MTOK * HIDN * 4;
  _Float16* x16  = (_Float16*)p;  p += (size_t)MTOK * FFN  * 2;
  _Float16* q16  = (_Float16*)p;  p += (size_t)MTOK * HIDN * 2;
  _Float16* k16  = (_Float16*)p;  p += (size_t)MTOK * HIDN * 2;
  _Float16* v16  = (_Float16*)p;  p += (size_t)MTOK * HIDN * 2;
  _Float16* w16  = (_Float16*)p;  p += (size_t)FFN * HIDN * 2;

  const int HH = HIDN * HIDN, HI = HIDN * FFN;
  const int nH = MTOK * HIDN, nF = MTOK * FFN;
  dim3 gN(HIDN / 64, MTOK / 128);          // GEMM to 768 cols
  dim3 gF(FFN  / 64, MTOK / 128);          // GEMM to 3072 cols
  dim3 tHH(HIDN / 32, HIDN / 32);          // transpose grids
  dim3 tHI(FFN / 32,  HIDN / 32);
  dim3 tIH(HIDN / 32, FFN / 32);

  mfp_embed_ln<<<MTOK, 256, 0, stream>>>(emb, pos, tok, elng, elnb, h);

  for (int l = 0; l < NLAYERS; ++l) {
    mfp_f2h<<<nH / 256, 256, 0, stream>>>(h, x16, nH);
    // Q (scaled by 1/sqrt(64) into f16), K, V
    mfp_wtrans<<<tHH, 256, 0, stream>>>(Wq + (size_t)l * HH, w16, HIDN, HIDN);
    mfp_gemm<<<gN, 256, 0, stream>>>(x16, w16, bq + (size_t)l * HIDN, nullptr, q16,
                                     MTOK, HIDN, HIDN, 0, 0.125f);
    mfp_wtrans<<<tHH, 256, 0, stream>>>(Wk + (size_t)l * HH, w16, HIDN, HIDN);
    mfp_gemm<<<gN, 256, 0, stream>>>(x16, w16, bk + (size_t)l * HIDN, nullptr, k16,
                                     MTOK, HIDN, HIDN, 0, 1.0f);
    mfp_wtrans<<<tHH, 256, 0, stream>>>(Wv + (size_t)l * HH, w16, HIDN, HIDN);
    mfp_gemm<<<gN, 256, 0, stream>>>(x16, w16, bv + (size_t)l * HIDN, nullptr, v16,
                                     MTOK, HIDN, HIDN, 0, 1.0f);
    // band attention
    mfp_attn<<<Bz * NHEADS * NCHUNK * 4, 64, 0, stream>>>(q16, k16, v16, am, attn);
    // output projection + residual LN
    mfp_f2h<<<nH / 256, 256, 0, stream>>>(attn, x16, nH);
    mfp_wtrans<<<tHH, 256, 0, stream>>>(Wo + (size_t)l * HH, w16, HIDN, HIDN);
    mfp_gemm<<<gN, 256, 0, stream>>>(x16, w16, bo + (size_t)l * HIDN, tmp, nullptr,
                                     MTOK, HIDN, HIDN, 0, 1.0f);
    mfp_add_ln<<<MTOK, 256, 0, stream>>>(h, tmp, l1g + (size_t)l * HIDN, l1b + (size_t)l * HIDN);
    // FFN: gelu(h@Wi+bi)@Wf+bf, residual LN
    mfp_f2h<<<nH / 256, 256, 0, stream>>>(h, x16, nH);
    mfp_wtrans<<<tHI, 256, 0, stream>>>(Wi + (size_t)l * HI, w16, HIDN, FFN);
    mfp_gemm<<<gF, 256, 0, stream>>>(x16, w16, bi + (size_t)l * FFN, tmp, nullptr,
                                     MTOK, FFN, HIDN, 1, 1.0f);
    mfp_f2h<<<nF / 256, 256, 0, stream>>>(tmp, x16, nF);
    mfp_wtrans<<<tIH, 256, 0, stream>>>(Wf + (size_t)l * HI, w16, FFN, HIDN);
    mfp_gemm<<<gN, 256, 0, stream>>>(x16, w16, bf + (size_t)l * HIDN, attn, nullptr,
                                     MTOK, HIDN, FFN, 0, 1.0f);
    mfp_add_ln<<<MTOK, 256, 0, stream>>>(h, attn, l2g + (size_t)l * HIDN, l2b + (size_t)l * HIDN);
  }

  mfp_head<<<MTOK, 256, 0, stream>>>(h, Wp, bp, out);
  (void)in_sizes; (void)n_in; (void)out_size; (void)ws_size;
}
